// MultiheadAttention_7713761263758
// MI455X (gfx1250) — matrix operand order repaired and checked by reading, not compiled, not run on hardware
//
#include <hip/hip_runtime.h>

// Problem sizes (fixed by the reference)
#define DMODEL 1024
#define DK     64
#define NHEAD  16
#define SEQ    2048
#define BATCH  2
#define MROWS  4096   // BATCH*SEQ

typedef __attribute__((ext_vector_type(16))) __bf16          v16bf;
typedef __attribute__((ext_vector_type(8)))  float           v8f;
typedef __attribute__((ext_vector_type(4)))  unsigned short  u16x4;
typedef __attribute__((ext_vector_type(8)))  unsigned short  u16x8;
typedef __attribute__((ext_vector_type(16))) unsigned short  u16x16;

__device__ __forceinline__ unsigned short f2bf(float f) {
  unsigned int u = __builtin_bit_cast(unsigned int, f);
  u += 0x7fffu + ((u >> 16) & 1u);             // round-to-nearest-even
  return (unsigned short)(u >> 16);
}

__device__ __forceinline__ u16x16 cat16(u16x8 lo, u16x8 hi) {
  return __builtin_shufflevector(lo, hi, 0,1,2,3,4,5,6,7,8,9,10,11,12,13,14,15);
}

// D(16x16 f32) = A(16x32 bf16) x B(32x16 bf16) + C, f32 accumulate
__device__ __forceinline__ v8f wmma_bf16(u16x16 a, u16x16 b, v8f c) {
  return __builtin_amdgcn_wmma_f32_16x16x32_bf16(
      false, __builtin_bit_cast(v16bf, a),
      false, __builtin_bit_cast(v16bf, b),
      (short)0, c, false, false);
}

// CDNA5 async DMA: global (16B per lane) -> LDS, tracked by ASYNCcnt.
// lds generic pointer's low 32 bits are the LDS byte address.
__device__ __forceinline__ void async_g2l_b128(void* lds, const void* gptr) {
  unsigned lds_off = (unsigned)(unsigned long long)lds;
  asm volatile("global_load_async_to_lds_b128 %0, %1, off"
               :: "v"(lds_off), "v"(gptr) : "memory");
}
__device__ __forceinline__ void wait_async0() {
  asm volatile("s_wait_asynccnt 0" ::: "memory");
}

// ---------------------------------------------------------------- convert
__global__ __launch_bounds__(256)
void cvt_f32_bf16(const float* __restrict__ src, unsigned short* __restrict__ dst, int n4) {
  int i = blockIdx.x * blockDim.x + threadIdx.x;
  if (i < n4) {
    float4 v = ((const float4*)src)[i];
    u16x4 o = { f2bf(v.x), f2bf(v.y), f2bf(v.z), f2bf(v.w) };
    ((u16x4*)dst)[i] = o;
  }
}

// ---------------------------------------------------------------- GEMM
// C[M,N] = A[M,K](bf16) @ W[N,K]^T(bf16) + bias[N]
// Wave: 16M x 128N. Block (8 waves): 128M x 128N.
// W chunks DMA'd to LDS via async loads; A fragment double-buffered in registers.
__global__ __launch_bounds__(256)
void gemm_bf16(const unsigned short* __restrict__ A,
               const unsigned short* __restrict__ W,
               const float* __restrict__ bias,
               float* __restrict__ Cf,
               unsigned short* __restrict__ Cb,
               int M, int N, int K, int klin) {
  __shared__ __align__(16) unsigned short Wt[128][136]; // [n_local][k_local], padded

  const int lane   = threadIdx.x & 31;
  const int wave   = threadIdx.x >> 5;
  const int lane16 = lane & 15;
  const int hi     = lane >> 4;
  const int m0 = blockIdx.y * 128 + wave * 16;
  const int n0 = blockIdx.x * 128;

  v8f acc[8];
#pragma unroll
  for (int t = 0; t < 8; ++t) acc[t] = (v8f){0,0,0,0,0,0,0,0};

  const unsigned short* arow = A + (size_t)(m0 + lane16) * K;

  // A fragment double buffer (lanes 0-15: K {+0..7,+16..23}; lanes 16-31: {+8,+24})
  u16x16 av[2];
  av[0] = cat16(*(const u16x8*)(arow + hi * 8), *(const u16x8*)(arow + 16 + hi * 8));

  for (int k0 = 0; k0 < K; k0 += 128) {
    __syncthreads();
    // async DMA of W[n0..n0+128)[k0..k0+128) -> LDS (no VGPR round trip)
#pragma unroll
    for (int i = 0; i < 8; ++i) {
      const int i8 = (int)threadIdx.x + i * 256;      // 2048 16B transfers
      const int r = i8 >> 4, c = (i8 & 15) * 8;
      async_g2l_b128(&Wt[r][c], W + (size_t)(n0 + r) * K + k0 + c);
    }
    wait_async0();
    __syncthreads();

#pragma unroll
    for (int kk = 0; kk < 128; kk += 32) {
      const int cur = (kk >> 5) & 1, nxt = cur ^ 1;
      // issue next A-fragment global load; it completes under the WMMAs
      int kn = k0 + kk + 32; if (kn > K - 32) kn = K - 32;
      av[nxt] = cat16(*(const u16x8*)(arow + kn + hi * 8),
                      *(const u16x8*)(arow + kn + 16 + hi * 8));
      // gather all 8 B fragments from LDS, then run the 8 WMMAs back-to-back
      u16x16 bf[8];
#pragma unroll
      for (int t = 0; t < 8; ++t) {
        const unsigned short* p = &Wt[t * 16 + lane16][kk + hi * 8];
        bf[t] = cat16(*(const u16x8*)p, *(const u16x8*)(p + 16));
      }
      __builtin_amdgcn_sched_barrier(0);
#pragma unroll
      for (int t = 0; t < 8; ++t) acc[t] = wmma_bf16(av[cur], bf[t], acc[t]);
    }
  }

#pragma unroll
  for (int t = 0; t < 8; ++t) {
    const int col = n0 + t * 16 + lane16;
    const float bb = bias[col];
#pragma unroll
    for (int v = 0; v < 8; ++v) {
      const int row = m0 + v + hi * 8;
      const float val = acc[t][v] + bb;
      if (Cb) Cb[(size_t)row * N + col] = f2bf(val);
      if (Cf) {
        size_t idx;
        if (klin) {   // k_lin[b, r*16 + col/64, h*64 + col%64] = Kp[b, 128h+r, col]
          const int b = row >> 11, s = row & 2047;
          const int h = s >> 7,   r = s & 127;
          const int s2 = r * 16 + (col >> 6);
          idx = ((size_t)(b * SEQ + s2)) * DMODEL + h * 64 + (col & 63);
        } else {
          idx = (size_t)row * N + col;
        }
        Cf[idx] = val;
      }
    }
  }
}

// ---------------------------------------------------------------- flash attention
// Per (qt, h, b) block: 8 waves, wave owns 16 query rows of a 128-row q tile.
// Head slab = contiguous rows [128h, 128h+128) of [B,S,D] viewed as [2048][64].
// K tile and transposed V tile share one LDS buffer (disjoint live ranges).
__global__ __launch_bounds__(256)
void flash_attn(const unsigned short* __restrict__ Qp,
                const unsigned short* __restrict__ Kp,
                const unsigned short* __restrict__ Vp,
                unsigned short* __restrict__ O) {
  __shared__ __align__(16) unsigned short Pbuf[8][16][136]; // per-wave P, padded rows
  __shared__ __align__(16) unsigned short KVbuf[9216];      // K:[128][72] / Vt:[64][136]

  const int qt = blockIdx.x, h = blockIdx.y, b = blockIdx.z;
  const int lane   = threadIdx.x & 31;
  const int wave   = threadIdx.x >> 5;
  const int lane16 = lane & 15;
  const int hi     = lane >> 4;

  const size_t base = ((size_t)(b * SEQ + h * 128)) * DMODEL; // slab base (elements)
  const unsigned short* Qh = Qp + base;
  const unsigned short* Kh = Kp + base;
  const unsigned short* Vh = Vp + base;
  const int q0 = qt * 128 + wave * 16;

  // Q fragments for this wave's 16 rows, d split into 2 k-chunks of 32
  u16x16 qa[2];
#pragma unroll
  for (int c = 0; c < 2; ++c) {
    const unsigned short* r = Qh + (size_t)(q0 + lane16) * DK + c * 32;
    qa[c] = cat16(*(const u16x8*)(r + hi * 8), *(const u16x8*)(r + 16 + hi * 8));
  }

  float m_i[8], l_i[8];
#pragma unroll
  for (int v = 0; v < 8; ++v) { m_i[v] = -3.0e38f; l_i[v] = 0.0f; }
  v8f o_acc[4];
#pragma unroll
  for (int t = 0; t < 4; ++t) o_acc[t] = (v8f){0,0,0,0,0,0,0,0};

  for (int kt = 0; kt <= qt; ++kt) {
    // ---- async DMA K tile [128 keys][64 dk] -> KVbuf rows of 72 halves
    __syncthreads();
#pragma unroll
    for (int i = 0; i < 4; ++i) {
      const int i8 = (int)threadIdx.x + i * 256;      // 1024 16B transfers
      const int r = i8 >> 3, c = (i8 & 7) * 8;
      async_g2l_b128(&KVbuf[r * 72 + c], Kh + (size_t)(kt * 128 + r) * DK + c);
    }
    // prefetch next key tile while this one is consumed
    if (kt < qt)
      __builtin_prefetch(Kh + (size_t)((kt + 1) * 128) * DK + (int)threadIdx.x * 32, 0, 1);
    wait_async0();
    __syncthreads();

    // ---- S = Q @ K^T : 4 groups of 2 key n-tiles (frag loads batched per group)
    v8f s[8];
#pragma unroll
    for (int g = 0; g < 4; ++g) {
      u16x16 kf[2][2];
#pragma unroll
      for (int u = 0; u < 2; ++u) {
#pragma unroll
        for (int c = 0; c < 2; ++c) {
          const unsigned short* kr =
              &KVbuf[((g * 2 + u) * 16 + lane16) * 72 + c * 32 + hi * 8];
          kf[u][c] = cat16(*(const u16x8*)kr, *(const u16x8*)(kr + 16));
        }
      }
      __builtin_amdgcn_sched_barrier(0);
#pragma unroll
      for (int u = 0; u < 2; ++u) {
        v8f a = (v8f){0,0,0,0,0,0,0,0};
        a = wmma_bf16(qa[0], kf[u][0], a);
        a = wmma_bf16(qa[1], kf[u][1], a);
        s[g * 2 + u] = a * 0.125f;  // 1/sqrt(64)
      }
    }

    if (kt == qt) {       // causal mask on the diagonal tile
#pragma unroll
      for (int t = 0; t < 8; ++t) {
        const int key = kt * 128 + t * 16 + lane16;
#pragma unroll
        for (int v = 0; v < 8; ++v) {
          const int q = q0 + v + hi * 8;
          if (key > q) s[t][v] = -1.0e30f;
        }
      }
    }

    // ---- online softmax (cross-lane butterfly within each 16-lane half)
    float mt[8];
#pragma unroll
    for (int v = 0; v < 8; ++v) {
      float mx = s[0][v];
#pragma unroll
      for (int t = 1; t < 8; ++t) mx = fmaxf(mx, s[t][v]);
      mt[v] = mx;
    }
#pragma unroll
    for (int d = 1; d < 16; d <<= 1) {
#pragma unroll
      for (int v = 0; v < 8; ++v) mt[v] = fmaxf(mt[v], __shfl_xor(mt[v], d, 32));
    }

    float mnew[8], scl[8], rs[8];
#pragma unroll
    for (int v = 0; v < 8; ++v) {
      mnew[v] = fmaxf(m_i[v], mt[v]);
      scl[v]  = __expf(m_i[v] - mnew[v]);
      rs[v]   = 0.0f;
    }

    // P = exp(S - m) -> bf16 into per-wave LDS, [row][key] layout
#pragma unroll
    for (int t = 0; t < 8; ++t) {
#pragma unroll
      for (int v = 0; v < 8; ++v) {
        const float p = __expf(s[t][v] - mnew[v]);
        rs[v] += p;
        Pbuf[wave][v + hi * 8][t * 16 + lane16] = f2bf(p);
      }
    }
#pragma unroll
    for (int d = 1; d < 16; d <<= 1) {
#pragma unroll
      for (int v = 0; v < 8; ++v) rs[v] += __shfl_xor(rs[v], d, 32);
    }
#pragma unroll
    for (int v = 0; v < 8; ++v) { l_i[v] = l_i[v] * scl[v] + rs[v]; m_i[v] = mnew[v]; }
#pragma unroll
    for (int t = 0; t < 4; ++t) {
#pragma unroll
      for (int v = 0; v < 8; ++v) o_acc[t][v] *= scl[v];
    }

    // ---- stage V tile transposed [dk][key]: vector global reads, b16 LDS scatter
    __syncthreads();   // all waves done reading K fragments
#pragma unroll
    for (int i = 0; i < 4; ++i) {
      const int i8 = (int)threadIdx.x + i * 256;      // 1024 chunks of 8 dk values
      const int key = i8 >> 3, c8 = (i8 & 7) * 8;
      u16x8 vv = *(const u16x8*)(Vh + (size_t)(kt * 128 + key) * DK + c8);
#pragma unroll
      for (int e = 0; e < 8; ++e) KVbuf[(c8 + e) * 136 + key] = vv[e];
    }
    __syncthreads();

    // ---- O += P @ V  (4 key-chunks of 32, 4 dk n-tiles; frags batched)
#pragma unroll
    for (int c = 0; c < 4; ++c) {
      const unsigned short* pr = &Pbuf[wave][lane16][c * 32];
      u16x16 pa = cat16(*(const u16x8*)(pr + hi * 8), *(const u16x8*)(pr + 16 + hi * 8));
      u16x16 vf[4];
#pragma unroll
      for (int t = 0; t < 4; ++t) {
        const unsigned short* vr = &KVbuf[(t * 16 + lane16) * 136 + c * 32 + hi * 8];
        vf[t] = cat16(*(const u16x8*)vr, *(const u16x8*)(vr + 16));
      }
      __builtin_amdgcn_sched_barrier(0);
#pragma unroll
      for (int t = 0; t < 4; ++t) o_acc[t] = wmma_bf16(pa, vf[t], o_acc[t]);
    }
  }

  // epilogue: normalize and write O[b, s2, h*64 + dk] (bf16 for the final GEMM)
#pragma unroll
  for (int t = 0; t < 4; ++t) {
#pragma unroll
    for (int v = 0; v < 8; ++v) {
      const int row = q0 + v + hi * 8;
      const float val = o_acc[t][v] / l_i[v];
      O[(size_t)(b * SEQ + row) * DMODEL + h * 64 + t * 16 + lane16] = f2bf(val);
    }
  }
}

// ---------------------------------------------------------------- launcher
extern "C" void kernel_launch(void* const* d_in, const int* in_sizes, int n_in,
                              void* d_out, int out_size, void* d_ws, size_t ws_size,
                              hipStream_t stream) {
  (void)in_sizes; (void)n_in; (void)out_size; (void)ws_size;
  const float* Q  = (const float*)d_in[0];
  const float* K  = (const float*)d_in[1];
  const float* V  = (const float*)d_in[2];
  // d_in[3] = mask: causal, computed analytically in-kernel
  const float* wq = (const float*)d_in[4];
  const float* bq = (const float*)d_in[5];
  const float* wk = (const float*)d_in[6];
  const float* bk = (const float*)d_in[7];
  const float* wv = (const float*)d_in[8];
  const float* bv = (const float*)d_in[9];
  const float* wo = (const float*)d_in[10];
  const float* bo = (const float*)d_in[11];

  const size_t ACT = (size_t)MROWS * DMODEL;   // 4,194,304 elements
  const size_t WEL = (size_t)DMODEL * DMODEL;  // 1,048,576 elements

  unsigned short* w   = (unsigned short*)d_ws;
  unsigned short* Qbf = w;
  unsigned short* Kbf = Qbf + ACT;
  unsigned short* Vbf = Kbf + ACT;
  unsigned short* Wq  = Vbf + ACT;
  unsigned short* Wk  = Wq + WEL;
  unsigned short* Wv  = Wk + WEL;
  unsigned short* Wo  = Wv + WEL;
  unsigned short* Qp  = Wo + WEL;
  unsigned short* Kp  = Qp + ACT;
  unsigned short* Vp  = Kp + ACT;
  unsigned short* Ob  = Vp + ACT;

  // f32 -> bf16 staging
  {
    int n4 = (int)(ACT / 4);
    cvt_f32_bf16<<<(n4 + 255) / 256, 256, 0, stream>>>(Q, Qbf, n4);
    cvt_f32_bf16<<<(n4 + 255) / 256, 256, 0, stream>>>(K, Kbf, n4);
    cvt_f32_bf16<<<(n4 + 255) / 256, 256, 0, stream>>>(V, Vbf, n4);
    int w4 = (int)(WEL / 4);
    cvt_f32_bf16<<<(w4 + 255) / 256, 256, 0, stream>>>(wq, Wq, w4);
    cvt_f32_bf16<<<(w4 + 255) / 256, 256, 0, stream>>>(wk, Wk, w4);
    cvt_f32_bf16<<<(w4 + 255) / 256, 256, 0, stream>>>(wv, Wv, w4);
    cvt_f32_bf16<<<(w4 + 255) / 256, 256, 0, stream>>>(wo, Wo, w4);
  }

  dim3 gg(DMODEL / 128, MROWS / 128);
  // Q/K/V projections; K writes k_lin (permuted f32) straight into d_out[0..ACT)
  gemm_bf16<<<gg, 256, 0, stream>>>(Qbf, Wq, bq, nullptr,       Qp, MROWS, DMODEL, DMODEL, 0);
  gemm_bf16<<<gg, 256, 0, stream>>>(Kbf, Wk, bk, (float*)d_out, Kp, MROWS, DMODEL, DMODEL, 1);
  gemm_bf16<<<gg, 256, 0, stream>>>(Vbf, Wv, bv, nullptr,       Vp, MROWS, DMODEL, DMODEL, 0);

  flash_attn<<<dim3(SEQ / 128, NHEAD, BATCH), 256, 0, stream>>>(Qp, Kp, Vp, Ob);

  // v_att = O @ w_o^T + b_o -> f32 second half of d_out
  gemm_bf16<<<gg, 256, 0, stream>>>(Ob, Wo, bo, (float*)d_out + ACT, nullptr,
                                    MROWS, DMODEL, DMODEL, 0);
}
